// ParallelGroupedMLP_2302102471522
// MI455X (gfx1250) — matrix operand order repaired and checked by reading, not compiled, not run on hardware
//
#include <hip/hip_runtime.h>

typedef __attribute__((ext_vector_type(16))) __bf16 v16bf;
typedef __attribute__((ext_vector_type(2)))  __bf16 v2bf;
typedef __attribute__((ext_vector_type(8)))  float  v8f;

#define E_     8
#define T_     512
#define H_     1024
#define F_     4096
#define R_     4
#define HN_    1280
#define N_     4096

#define BM 128
#define BN 128
#define BK 64
#define LSTR 72   // LDS row stride in bf16 elems (64 + 8 pad)

// hardware f32->bf16 (RNE via v_cvt); packed pair -> one dword
static __device__ __forceinline__ unsigned short f2bf(float f) {
  union { __bf16 b; unsigned short u; } cv;
  cv.b = (__bf16)f;
  return cv.u;
}
static __device__ __forceinline__ unsigned pack2bf(float lo, float hi) {
  union { v2bf v; unsigned u; } cv;
  cv.v.x = (__bf16)lo;
  cv.v.y = (__bf16)hi;
  return cv.u;
}
static __device__ __forceinline__ float bf2f(unsigned short h) {
  return __uint_as_float(((unsigned)h) << 16);
}

struct Frag { union { v16bf v; uint4 q[2]; }; };

// ---- fragment loads from LDS (ISA 7.12.2 layouts) ----
// A 16x32 bf16: lane<16 -> row=lane, K in {0..7, 16..23}; lane>=16 -> K in {8..15, 24..31}
static __device__ __forceinline__ void ldFragA(Frag& f, const unsigned short* lds,
                                               int rowBase, int ks, int lane) {
  const unsigned short* p = lds + (rowBase + (lane & 15)) * LSTR + ks + ((lane >> 4) << 3);
  f.q[0] = *(const uint4*)(p);
  f.q[1] = *(const uint4*)(p + 16);
}
// B 32x16 bf16: lane<16 -> col=lane, K in {0..7, 16..23}; lane>=16 -> K in {8..15, 24..31} (LDS B stored [n][k])
static __device__ __forceinline__ void ldFragB(Frag& f, const unsigned short* lds,
                                               int colBase, int ks, int lane) {
  const unsigned short* p = lds + (colBase + (lane & 15)) * LSTR + ks + ((lane >> 4) << 3);
  f.q[0] = *(const uint4*)(p);
  f.q[1] = *(const uint4*)(p + 16);
}

// one BK=64 chunk: two k-steps of v_wmma_f32_16x16x32_bf16, 8 frags/wave (64x32 per wave)
static __device__ __forceinline__ void mmaChunk(const unsigned short* lA, const unsigned short* lB,
                                                int waveM, int waveN, int lane, v8f acc[4][2]) {
#pragma unroll
  for (int ks = 0; ks < BK; ks += 32) {
    Frag a[4], b[2];
#pragma unroll
    for (int i = 0; i < 4; i++) ldFragA(a[i], lA, waveM * 64 + i * 16, ks, lane);
#pragma unroll
    for (int j = 0; j < 2; j++) ldFragB(b[j], lB, waveN * 32 + j * 16, ks, lane);
#pragma unroll
    for (int i = 0; i < 4; i++)
#pragma unroll
      for (int j = 0; j < 2; j++)
        acc[i][j] = __builtin_amdgcn_wmma_f32_16x16x32_bf16(
            false, a[i].v, false, b[j].v, (short)0, acc[i][j], false, false);
  }
}

// ---- tile loaders ----
static __device__ __forceinline__ void ldA_bf16(unsigned short* lA, const unsigned short* src,
                                                size_t ld, int row0, int k0) {
  int tid = threadIdx.x;
#pragma unroll
  for (int i = 0; i < 4; i++) {
    int idx = tid + i * 256, row = idx >> 3, seg = idx & 7;
    uint4 v = *(const uint4*)(src + (size_t)(row0 + row) * ld + k0 + seg * 8);
    *(uint4*)(lA + row * LSTR + seg * 8) = v;
  }
}
// fp32 global (row-major [K][ld]) -> LDS B [n][k], packed 2 bf16 per dword store
static __device__ __forceinline__ void ldB_f32(unsigned short* lB, const float* src,
                                               size_t ld, int k0, int n0) {
  int tid = threadIdx.x;
#pragma unroll
  for (int i = 0; i < 16; i++) {
    int idx = tid + i * 256, n = idx & 127, kp = idx >> 7;  // kp = 0..31 (k pairs)
    float f0 = src[(size_t)(k0 + 2 * kp) * ld + n0 + n];
    float f1 = src[(size_t)(k0 + 2 * kp + 1) * ld + n0 + n];
    *(unsigned*)(lB + n * LSTR + 2 * kp) = pack2bf(f0, f1);
  }
}
static __device__ __forceinline__ void ldB_bf16(unsigned short* lB, const unsigned short* src,
                                                size_t ld, int k0, int n0) {
  int tid = threadIdx.x;
#pragma unroll
  for (int i = 0; i < 16; i++) {
    int idx = tid + i * 256, n = idx & 127, kp = idx >> 7;
    unsigned s0 = src[(size_t)(k0 + 2 * kp) * ld + n0 + n];
    unsigned s1 = src[(size_t)(k0 + 2 * kp + 1) * ld + n0 + n];
    *(unsigned*)(lB + n * LSTR + 2 * kp) = s0 | (s1 << 16);
  }
}
// generate Px[n, p=k*4+r] = x_bf16[n,k] * low[n,r] for p-chunk [p0, p0+64)
static __device__ __forceinline__ void ldA_px(unsigned short* lA, const unsigned short* xb,
                                              int tok0, int p0, const float* lowT) {
  int tid = threadIdx.x;
#pragma unroll
  for (int i = 0; i < 4; i++) {
    int idx = tid + i * 256, row = idx >> 3, seg = idx & 7;
    int pb = p0 + seg * 8;
    unsigned w[4];
#pragma unroll
    for (int q = 0; q < 4; q++) {
      int pA = pb + 2 * q, pB = pA + 1;
      float xa = bf2f(xb[(size_t)(tok0 + row) * H_ + (pA >> 2)]);
      float xc = bf2f(xb[(size_t)(tok0 + row) * H_ + (pB >> 2)]);
      w[q] = pack2bf(xa * lowT[row * 4 + (pA & 3)], xc * lowT[row * 4 + (pB & 3)]);
    }
    uint4 v; v.x = w[0]; v.y = w[1]; v.z = w[2]; v.w = w[3];
    *(uint4*)(lA + row * LSTR + seg * 8) = v;
  }
}

// double-buffered GEMM main loop: one barrier per chunk (next-chunk loads issued
// before current-chunk WMMAs so global traffic hides under the matrix ops)
template <typename LA, typename LB>
static __device__ __forceinline__ void gemmPipe(unsigned short (*lA)[BM * LSTR],
                                                unsigned short (*lB)[BN * LSTR],
                                                int chunks, LA loadA, LB loadB,
                                                int waveM, int waveN, int lane, v8f acc[4][2]) {
  loadA(lA[0], 0);
  loadB(lB[0], 0);
  __syncthreads();
#pragma unroll 1
  for (int c = 0; c < chunks; c++) {
    int cur = c & 1;
    if (c + 1 < chunks) {
      loadA(lA[cur ^ 1], c + 1);
      loadB(lB[cur ^ 1], c + 1);
    }
    mmaChunk(lA[cur], lB[cur], waveM, waveN, lane, acc);
    __syncthreads();
  }
}

// ================= small setup kernels =================
__global__ __launch_bounds__(256) void k_sum_contig(const float* __restrict__ W, float* __restrict__ out,
                                                    int inner, int rowlen, int ldout, int outoff) {
  __shared__ float red[256];
  int row = blockIdx.x, tid = threadIdx.x;
  const float* p = W + (size_t)row * rowlen;
  float s = 0.f;
  for (int i = tid; i < rowlen; i += 256) s += p[i];
  red[tid] = s; __syncthreads();
  for (int off = 128; off > 0; off >>= 1) { if (tid < off) red[tid] += red[tid + off]; __syncthreads(); }
  if (tid == 0) { int e = row / inner, j = row % inner; out[(size_t)e * ldout + outoff + j] = red[0]; }
}
__global__ __launch_bounds__(256) void k_sum_strided(const float* __restrict__ W, float* __restrict__ out,
                                                     int rows, int cols, int ldout, int outoff) {
  int gid = blockIdx.x * 256 + threadIdx.x;
  int e = gid / cols, j = gid % cols;
  const float* p = W + (size_t)e * rows * cols + j;
  float s = 0.f;
  for (int i = 0; i < rows; i++) s += p[(size_t)i * cols];
  out[(size_t)e * ldout + outoff + j] = s;
}
__global__ __launch_bounds__(128) void k_embc(const float* __restrict__ sig1, const float* __restrict__ sig2,
                                              const float* __restrict__ w1_emb, const float* __restrict__ w2_emb,
                                              float* __restrict__ embc) {
  int e = blockIdx.x >> 1, which = blockIdx.x & 1, b = threadIdx.x;
  const float* sig = (which ? sig2 : sig1) + (size_t)e * (H_ + F_);
  const float* wemb = which ? w2_emb : w1_emb;
  float s = 0.f;
  for (int j = 0; j < H_ + F_; j++) s += sig[j] * wemb[(size_t)j * 128 + b];
  embc[e * 256 + which * 128 + b] = s;
}
// out[e, j] = sum_k embc[e,k] * W[(H+k)*width + j]   (expert tail folded into biases)
__global__ __launch_bounds__(256) void k_tailproj(const float* __restrict__ embc, const float* __restrict__ W,
                                                  float* __restrict__ out, int width) {
  int gid = blockIdx.x * 256 + threadIdx.x;
  int e = gid / width, j = gid % width;
  const float* w = W + (size_t)H_ * width + j;
  const float* c = embc + e * 256;
  float s = 0.f;
  for (int k = 0; k < 256; k++) s += c[k] * w[(size_t)k * width];
  out[(size_t)e * width + j] = s;
}
__global__ __launch_bounds__(256) void k_cvt(const float* __restrict__ src, unsigned short* __restrict__ dst) {
  size_t base = ((size_t)blockIdx.x * 256 + threadIdx.x) * 4;
  float4 f = *(const float4*)(src + base);
  uint2 o; o.x = pack2bf(f.x, f.y); o.y = pack2bf(f.z, f.w);
  *(uint2*)(dst + base) = o;
}
// dst[(k*4+r)*W + j] = bf16(src[k*4W + r*W + j])
__global__ __launch_bounds__(256) void k_reorder(const float* __restrict__ src, unsigned short* __restrict__ dst, int W) {
  size_t base = ((size_t)blockIdx.x * 256 + threadIdx.x) * 4;
#pragma unroll
  for (int t = 0; t < 4; t++) {
    size_t idx = base + t;
    int p = (int)(idx / W), j = (int)(idx % W);
    dst[idx] = f2bf(src[(size_t)(p >> 2) * 4 * W + (size_t)(p & 3) * W + j]);
  }
}
// A2t[f*4096 + (k*4+r)] = bf16(A2[k*16384 + f*4 + r]),  k < 1024
__global__ __launch_bounds__(256) void k_transposeA2(const float* __restrict__ A2, unsigned short* __restrict__ dst) {
  size_t base = ((size_t)blockIdx.x * 256 + threadIdx.x) * 4;
#pragma unroll
  for (int t = 0; t < 4; t++) {
    size_t idx = base + t;
    int f = (int)(idx >> 12), p = (int)(idx & 4095);
    dst[idx] = f2bf(A2[(size_t)(p >> 2) * 16384 + (size_t)f * 4 + (p & 3)]);
  }
}
// low2[n,r] += sum_f a[n,f] * A2c[e, f*4+r]
__global__ __launch_bounds__(256) void k_low2tail(const unsigned short* __restrict__ a,
                                                  const float* __restrict__ A2c, float* __restrict__ low2) {
  __shared__ float red[256 * 4];
  int n = blockIdx.x, tid = threadIdx.x, e = n >> 9;
  const unsigned short* ar = a + (size_t)n * F_;
  const float* c = A2c + (size_t)e * (F_ * R_);
  float s0 = 0, s1 = 0, s2 = 0, s3 = 0;
  for (int f = tid; f < F_; f += 256) {
    float av = bf2f(ar[f]);
    s0 += av * c[f * 4 + 0]; s1 += av * c[f * 4 + 1];
    s2 += av * c[f * 4 + 2]; s3 += av * c[f * 4 + 3];
  }
  red[tid * 4 + 0] = s0; red[tid * 4 + 1] = s1; red[tid * 4 + 2] = s2; red[tid * 4 + 3] = s3;
  __syncthreads();
  for (int off = 128; off > 0; off >>= 1) {
    if (tid < off) {
#pragma unroll
      for (int r = 0; r < 4; r++) red[tid * 4 + r] += red[(tid + off) * 4 + r];
    }
    __syncthreads();
  }
  if (tid < 4) atomicAdd(&low2[(size_t)n * 4 + tid], red[tid]);
}

// ================= WMMA kernels =================

// K1: Y1A = xb @ A1[:H] (+bias1A[e]); epilogue low1[n,r] += Y1A[n,h*4+r]*x[n,h]
__global__ __launch_bounds__(256) void k1_y1a_low(const unsigned short* __restrict__ xb,
                                                  const float* __restrict__ A1,
                                                  const float* __restrict__ bias1A,
                                                  const float* __restrict__ x,
                                                  float* __restrict__ low1) {
  __shared__ __align__(16) unsigned short lA[2][BM * LSTR];
  __shared__ __align__(16) unsigned short lB[2][BN * LSTR];
  __shared__ float lowAcc[BM * 4];
  int tid = threadIdx.x, lane = tid & 31, wave = tid >> 5, waveM = wave >> 2, waveN = wave & 3;
  int tok0 = blockIdx.x * BM, n0 = blockIdx.y * BN, e = tok0 >> 9;
  for (int i = tid; i < BM * 4; i += 256) lowAcc[i] = 0.f;
  v8f zero = {};
  v8f acc[4][2];
#pragma unroll
  for (int i = 0; i < 4; i++) for (int j = 0; j < 2; j++) acc[i][j] = zero;
  gemmPipe(lA, lB, H_ / BK,
           [&](unsigned short* d, int c) { ldA_bf16(d, xb, H_, tok0, c * BK); },
           [&](unsigned short* d, int c) { ldB_f32(d, A1, H_ * R_, c * BK, n0); },
           waveM, waveN, lane, acc);
  const float* bA = bias1A + (size_t)e * (H_ * R_);
#pragma unroll
  for (int i = 0; i < 4; i++)
#pragma unroll
    for (int j = 0; j < 2; j++)
#pragma unroll
      for (int v = 0; v < 8; v++) {
        int m = waveM * 64 + i * 16 + ((lane >> 4) << 3) + v;
        int n = waveN * 32 + j * 16 + (lane & 15);
        int col = n0 + n;
        float val = acc[i][j][v] + bA[col];
        float xv = x[(size_t)(tok0 + m) * H_ + (col >> 2)];
        atomicAdd(&lowAcc[m * 4 + (col & 3)], val * xv);
      }
  __syncthreads();
  for (int i = tid; i < BM * 4; i += 256) atomicAdd(&low1[(size_t)tok0 * 4 + i], lowAcc[i]);
}

// K2: h = x@w1[e] + Px@B1x' + low·bias1B[e];  a = gelu(h) -> bf16
__global__ __launch_bounds__(256) void k2_h_gelu(const unsigned short* __restrict__ xb,
                                                 const float* __restrict__ w1,
                                                 const unsigned short* __restrict__ B1xb,
                                                 const float* __restrict__ low1,
                                                 const float* __restrict__ bias1B,
                                                 unsigned short* __restrict__ aout) {
  __shared__ __align__(16) unsigned short lA[2][BM * LSTR];
  __shared__ __align__(16) unsigned short lB[2][BN * LSTR];
  __shared__ float lowT[BM * 4];
  int tid = threadIdx.x, lane = tid & 31, wave = tid >> 5, waveM = wave >> 2, waveN = wave & 3;
  int tok0 = blockIdx.x * BM, n0 = blockIdx.y * BN, e = tok0 >> 9;
  for (int i = tid; i < BM * 4; i += 256) lowT[i] = low1[(size_t)tok0 * 4 + i];
  v8f zero = {};
  v8f acc[4][2];
#pragma unroll
  for (int i = 0; i < 4; i++) for (int j = 0; j < 2; j++) acc[i][j] = zero;
  const float* w1e = w1 + (size_t)e * H_ * F_;
  gemmPipe(lA, lB, H_ / BK,                         // base GEMM
           [&](unsigned short* d, int c) { ldA_bf16(d, xb, H_, tok0, c * BK); },
           [&](unsigned short* d, int c) { ldB_f32(d, w1e, F_, c * BK, n0); },
           waveM, waveN, lane, acc);
  gemmPipe(lA, lB, (H_ * R_) / BK,                  // LoRA delta via Px
           [&](unsigned short* d, int c) { ldA_px(d, xb, tok0, c * BK, lowT); },
           [&](unsigned short* d, int c) { ldB_bf16(d, B1xb, F_, c * BK, n0); },
           waveM, waveN, lane, acc);
  const float* bB = bias1B + (size_t)e * (F_ * R_);
#pragma unroll
  for (int i = 0; i < 4; i++)
#pragma unroll
    for (int j = 0; j < 2; j++)
#pragma unroll
      for (int v = 0; v < 8; v++) {
        int m = waveM * 64 + i * 16 + ((lane >> 4) << 3) + v;
        int n = waveN * 32 + j * 16 + (lane & 15);
        int col = n0 + n;
        float h = acc[i][j][v]
                + lowT[m * 4 + 0] * bB[0 * F_ + col] + lowT[m * 4 + 1] * bB[1 * F_ + col]
                + lowT[m * 4 + 2] * bB[2 * F_ + col] + lowT[m * 4 + 3] * bB[3 * F_ + col];
        float a = 0.5f * h * (1.f + erff(h * 0.70710678118f));   // exact gelu
        aout[(size_t)(tok0 + m) * F_ + col] = f2bf(a);
      }
}

// K3: Za = a @ A2t ; epilogue low2[n,r] += Za[n,k*4+r]*x[n,k]
__global__ __launch_bounds__(256) void k3_za_low2(const unsigned short* __restrict__ aact,
                                                  const unsigned short* __restrict__ A2tb,
                                                  const float* __restrict__ x,
                                                  float* __restrict__ low2) {
  __shared__ __align__(16) unsigned short lA[2][BM * LSTR];
  __shared__ __align__(16) unsigned short lB[2][BN * LSTR];
  __shared__ float lowAcc[BM * 4];
  int tid = threadIdx.x, lane = tid & 31, wave = tid >> 5, waveM = wave >> 2, waveN = wave & 3;
  int tok0 = blockIdx.x * BM, n0 = blockIdx.y * BN;
  for (int i = tid; i < BM * 4; i += 256) lowAcc[i] = 0.f;
  v8f zero = {};
  v8f acc[4][2];
#pragma unroll
  for (int i = 0; i < 4; i++) for (int j = 0; j < 2; j++) acc[i][j] = zero;
  gemmPipe(lA, lB, F_ / BK,
           [&](unsigned short* d, int c) { ldA_bf16(d, aact, F_, tok0, c * BK); },
           [&](unsigned short* d, int c) { ldB_bf16(d, A2tb, H_ * R_, c * BK, n0); },
           waveM, waveN, lane, acc);
#pragma unroll
  for (int i = 0; i < 4; i++)
#pragma unroll
    for (int j = 0; j < 2; j++)
#pragma unroll
      for (int v = 0; v < 8; v++) {
        int m = waveM * 64 + i * 16 + ((lane >> 4) << 3) + v;
        int n = waveN * 32 + j * 16 + (lane & 15);
        int col = n0 + n;
        float xv = x[(size_t)(tok0 + m) * H_ + (col >> 2)];
        atomicAdd(&lowAcc[m * 4 + (col & 3)], acc[i][j][v] * xv);
      }
  __syncthreads();
  for (int i = tid; i < BM * 4; i += 256) atomicAdd(&low2[(size_t)tok0 * 4 + i], lowAcc[i]);
}

// K4: out = a@w2[e] + P2x@B2x' + low2·bias2B[e]
__global__ __launch_bounds__(256) void k4_out(const unsigned short* __restrict__ aact,
                                              const float* __restrict__ w2,
                                              const unsigned short* __restrict__ xb,
                                              const unsigned short* __restrict__ B2xb,
                                              const float* __restrict__ low2,
                                              const float* __restrict__ bias2B,
                                              float* __restrict__ out) {
  __shared__ __align__(16) unsigned short lA[2][BM * LSTR];
  __shared__ __align__(16) unsigned short lB[2][BN * LSTR];
  __shared__ float lowT[BM * 4];
  int tid = threadIdx.x, lane = tid & 31, wave = tid >> 5, waveM = wave >> 2, waveN = wave & 3;
  int tok0 = blockIdx.x * BM, n0 = blockIdx.y * BN, e = tok0 >> 9;
  for (int i = tid; i < BM * 4; i += 256) lowT[i] = low2[(size_t)tok0 * 4 + i];
  v8f zero = {};
  v8f acc[4][2];
#pragma unroll
  for (int i = 0; i < 4; i++) for (int j = 0; j < 2; j++) acc[i][j] = zero;
  const float* w2e = w2 + (size_t)e * F_ * H_;
  gemmPipe(lA, lB, F_ / BK,                         // base GEMM2
           [&](unsigned short* d, int c) { ldA_bf16(d, aact, F_, tok0, c * BK); },
           [&](unsigned short* d, int c) { ldB_f32(d, w2e, H_, c * BK, n0); },
           waveM, waveN, lane, acc);
  gemmPipe(lA, lB, (H_ * R_) / BK,                  // LoRA delta 2 via P2x
           [&](unsigned short* d, int c) { ldA_px(d, xb, tok0, c * BK, lowT); },
           [&](unsigned short* d, int c) { ldB_bf16(d, B2xb, H_, c * BK, n0); },
           waveM, waveN, lane, acc);
  const float* bB = bias2B + (size_t)e * (H_ * R_);
#pragma unroll
  for (int i = 0; i < 4; i++)
#pragma unroll
    for (int j = 0; j < 2; j++)
#pragma unroll
      for (int v = 0; v < 8; v++) {
        int m = waveM * 64 + i * 16 + ((lane >> 4) << 3) + v;
        int n = waveN * 32 + j * 16 + (lane & 15);
        int col = n0 + n;
        float val = acc[i][j][v]
                  + lowT[m * 4 + 0] * bB[0 * H_ + col] + lowT[m * 4 + 1] * bB[1 * H_ + col]
                  + lowT[m * 4 + 2] * bB[2 * H_ + col] + lowT[m * 4 + 3] * bB[3 * H_ + col];
        out[(size_t)(tok0 + m) * H_ + col] = val;
      }
}

extern "C" void kernel_launch(void* const* d_in, const int* in_sizes, int n_in,
                              void* d_out, int out_size, void* d_ws, size_t ws_size,
                              hipStream_t stream) {
  (void)in_sizes; (void)n_in; (void)out_size; (void)ws_size;
  const float* x      = (const float*)d_in[0];
  const float* w1     = (const float*)d_in[2];
  const float* w2     = (const float*)d_in[3];
  const float* w1_emb = (const float*)d_in[4];
  const float* w2_emb = (const float*)d_in[5];
  const float* A1     = (const float*)d_in[6];
  const float* B1     = (const float*)d_in[7];
  const float* A2     = (const float*)d_in[8];
  const float* B2     = (const float*)d_in[9];

  size_t off = 0;
  auto carve = [&](size_t bytes) -> char* {
    char* p = (char*)d_ws + off; off += (bytes + 255) & ~(size_t)255; return p;
  };
  float* sig1   = (float*)carve((size_t)E_ * (H_ + F_) * 4);
  float* sig2   = (float*)carve((size_t)E_ * (H_ + F_) * 4);
  float* embc   = (float*)carve((size_t)E_ * 256 * 4);
  float* bias1A = (float*)carve((size_t)E_ * H_ * R_ * 4);
  float* bias1B = (float*)carve((size_t)E_ * F_ * R_ * 4);
  float* A2c    = (float*)carve((size_t)E_ * F_ * R_ * 4);
  float* bias2B = (float*)carve((size_t)E_ * H_ * R_ * 4);
  float* low1   = (float*)carve((size_t)N_ * 4 * 4);
  float* low2   = (float*)carve((size_t)N_ * 4 * 4);
  unsigned short* xb   = (unsigned short*)carve((size_t)N_ * H_ * 2);
  unsigned short* aact = (unsigned short*)carve((size_t)N_ * F_ * 2);
  unsigned short* B1xb = (unsigned short*)carve((size_t)H_ * R_ * F_ * 2);
  unsigned short* B2xb = (unsigned short*)carve((size_t)H_ * R_ * H_ * 2);
  unsigned short* A2tb = (unsigned short*)carve((size_t)F_ * H_ * R_ * 2);

  hipMemsetAsync(low1, 0, (size_t)N_ * 4 * 4, stream);
  hipMemsetAsync(low2, 0, (size_t)N_ * 4 * 4, stream);

  // expert signatures: w1_sig=[rowsum_F | colsum_H], w2_sig=[colsum_F | rowsum_H]
  k_sum_contig <<<E_ * H_, 256, 0, stream>>>(w1, sig1, H_, F_, H_ + F_, 0);
  k_sum_contig <<<E_ * F_, 256, 0, stream>>>(w2, sig2, F_, H_, H_ + F_, H_);
  k_sum_strided<<<(E_ * F_) / 256, 256, 0, stream>>>(w1, sig1, H_, F_, H_ + F_, H_);
  k_sum_strided<<<(E_ * H_) / 256, 256, 0, stream>>>(w2, sig2, F_, H_, H_ + F_, 0);
  k_embc<<<E_ * 2, 128, 0, stream>>>(sig1, sig2, w1_emb, w2_emb, embc);

  // fold shared per-expert embedding tail into biases
  k_tailproj<<<(E_ * H_ * R_) / 256, 256, 0, stream>>>(embc, A1, bias1A, H_ * R_);
  k_tailproj<<<(E_ * F_ * R_) / 256, 256, 0, stream>>>(embc, B1, bias1B, F_ * R_);
  k_tailproj<<<(E_ * F_ * R_) / 256, 256, 0, stream>>>(embc, A2, A2c, F_ * R_);
  k_tailproj<<<(E_ * H_ * R_) / 256, 256, 0, stream>>>(embc, B2, bias2B, H_ * R_);

  // bf16 conversions / layout reorders
  k_cvt        <<<(N_ * H_) / 1024, 256, 0, stream>>>(x, xb);
  k_reorder    <<<((size_t)H_ * R_ * F_) / 1024, 256, 0, stream>>>(B1, B1xb, F_);
  k_reorder    <<<((size_t)H_ * R_ * H_) / 1024, 256, 0, stream>>>(B2, B2xb, H_);
  k_transposeA2<<<((size_t)F_ * H_ * R_) / 1024, 256, 0, stream>>>(A2, A2tb);

  // WMMA pipeline
  k1_y1a_low<<<dim3(N_ / BM, (H_ * R_) / BN), 256, 0, stream>>>(xb, A1, bias1A, x, low1);
  k2_h_gelu <<<dim3(N_ / BM, F_ / BN),        256, 0, stream>>>(xb, w1, B1xb, low1, bias1B, aact);
  k3_za_low2<<<dim3(N_ / BM, (H_ * R_) / BN), 256, 0, stream>>>(aact, A2tb, x, low2);
  k_low2tail<<<N_, 256, 0, stream>>>(aact, A2c, low2);
  k4_out    <<<dim3(N_ / BM, H_ / BN),        256, 0, stream>>>(aact, w2, xb, B2xb, low2, bias2B, (float*)d_out);
}
